// MultiHeadAttention_82660940579375
// MI455X (gfx1250) — matrix operand order repaired and checked by reading, not compiled, not run on hardware
//
#include <hip/hip_runtime.h>
#include <hip/hip_bf16.h>

// ---------------------------------------------------------------------------
// MHA (H=8 fat heads, D=512, S=2048, B=2) for MI455X / gfx1250 (wave32, WMMA)
// bf16 data / f32 accumulate via v_wmma_f32_16x16x32_bf16.
// Weight tiles staged to LDS with GLOBAL_LOAD_ASYNC_TO_LDS_B128 (ASYNCcnt),
// double-buffered.  Fragment loads grouped ahead of the wmma chain so matrix
// ops issue back-to-back instead of serializing on one register block.
// ---------------------------------------------------------------------------

#define BATCH 2
#define SEQ   2048
#define DM    512
#define NH    8
#define HD    (NH*DM)           // 4096
#define SCALE 0.044194173824159216f   // 1/sqrt(512)

typedef __attribute__((ext_vector_type(16))) __bf16 bf16x16;
typedef __attribute__((ext_vector_type(8)))  float  f32x8;

// round-to-nearest-even fp32 -> bf16 (bit pattern)
__device__ __forceinline__ unsigned short f2bf(float f) {
  unsigned int u = __float_as_uint(f);
  u += 0x7FFFu + ((u >> 16) & 1u);
  return (unsigned short)(u >> 16);
}
__device__ __forceinline__ unsigned int pack2bf(float a, float b) {
  return (unsigned int)f2bf(a) | ((unsigned int)f2bf(b) << 16);
}
// Load a 16-element bf16 WMMA fragment as two 16-byte chunks.
__device__ __forceinline__ bf16x16 load2(const unsigned short* p0,
                                         const unsigned short* p1) {
  union { bf16x16 v; uint4 q[2]; } u;
  u.q[0] = *(const uint4*)p0;
  u.q[1] = *(const uint4*)p1;
  return u.v;
}

// ---- CDNA5 async global->LDS copy (ASYNCcnt) ------------------------------
// LDS byte offset = low 32 bits of the generic shared-aperture address
// (ISA §10.2).
__device__ __forceinline__ unsigned ldsOff(const void* p) {
  return (unsigned)(uintptr_t)p;
}
__device__ __forceinline__ void async_b128(const unsigned short* g, unsigned lds) {
  asm volatile("global_load_async_to_lds_b128 %0, %1, off"
               :: "v"(lds), "v"((unsigned long long)(uintptr_t)g)
               : "memory");
}
__device__ __forceinline__ void wait_async2() {
  asm volatile("s_wait_asynccnt 0x2" ::: "memory");
}
__device__ __forceinline__ void wait_async0() {
  asm volatile("s_wait_asynccnt 0x0" ::: "memory");
}
// Cooperative fill of a 64-row x 32-col bf16 tile (4 KB) by 128 threads:
// each thread moves 32 bytes with two async b128 ops.
__device__ __forceinline__ void fill_tile(const unsigned short* __restrict__ Wg,
                                          size_t ldw,
                                          unsigned short (*Bb)[32], int tid) {
  const int r = tid >> 1, half = tid & 1;
  const unsigned short* g = Wg + (size_t)r * ldw + half * 16;
  unsigned l = ldsOff(&Bb[r][half * 16]);
  async_b128(g, l);
  async_b128(g + 8, l + 16);
}

// ---------------------------------------------------------------------------
// K0: fp32 -> bf16 elementwise
// ---------------------------------------------------------------------------
__global__ void k_cvt_x(const float* __restrict__ x, unsigned short* __restrict__ xb, int n) {
  int i = blockIdx.x * blockDim.x + threadIdx.x;
  if (i < n) xb[i] = f2bf(x[i]);
}

// ---------------------------------------------------------------------------
// K1: transpose-convert fp32 [rows][cols] -> bf16 [cols][rows] (per matrix z)
// ---------------------------------------------------------------------------
__global__ void k_transpose_bf16(const float* __restrict__ in, unsigned short* __restrict__ out,
                                 int rows, int cols) {
  __shared__ float tile[32][33];
  int m = blockIdx.z;
  const float* A = in + (size_t)m * rows * cols;
  unsigned short* B = out + (size_t)m * rows * cols;
  int c0 = blockIdx.x * 32, r0 = blockIdx.y * 32;
  int tx = threadIdx.x, ty = threadIdx.y;       // 32 x 8
  #pragma unroll
  for (int i = 0; i < 32; i += 8)
    tile[ty + i][tx] = A[(size_t)(r0 + ty + i) * cols + c0 + tx];
  __syncthreads();
  #pragma unroll
  for (int i = 0; i < 32; i += 8)
    B[(size_t)(c0 + ty + i) * rows + r0 + tx] = f2bf(tile[tx][ty + i]);
}

// ---------------------------------------------------------------------------
// K2: QKV projection GEMM.  block = 128 thr (4 waves); waves split M, share
// the 64-col weight tile which is async-staged to LDS (double-buffered).
//   Q[s][e] = (x@Wq + bq)*scale  -> Qb row-major [b*H+h][s][e]
//   K[s][e] = (x@Wk + bk)        -> Kb row-major
//   V       = (x@Wv + bv)        -> Vt TRANSPOSED [b*H+h][e][s]
// grid: (SEQ/64, DM/64, 3*BATCH*NH)
// ---------------------------------------------------------------------------
__global__ __launch_bounds__(128)
void k_qkv_gemm(const unsigned short* __restrict__ xb,
                const unsigned short* __restrict__ WqT,
                const unsigned short* __restrict__ WkT,
                const unsigned short* __restrict__ WvT,
                const float* __restrict__ bq, const float* __restrict__ bk,
                const float* __restrict__ bv,
                unsigned short* __restrict__ Qb, unsigned short* __restrict__ Kb,
                unsigned short* __restrict__ Vt) {
  __shared__ __align__(16) unsigned short Bbuf[2][64][32];

  const int lane  = threadIdx.x & 31;
  const int wv    = threadIdx.x >> 5;
  const int mtx   = blockIdx.z / (BATCH * NH);      // 0=Q 1=K 2=V
  const int bh    = blockIdx.z % (BATCH * NH);
  const int h     = bh % NH;
  const int b     = bh / NH;
  const int Mbase = blockIdx.x * 64 + wv * 16;
  const int Nbase = blockIdx.y * 64;
  const int ln16  = lane & 15;
  const int hi    = lane >> 4;
  const int ahalf = hi * 8;     // A-fragment K sub-offset / C-fragment row offset
  const int bhalf = hi * 16;    // unused; B fragment takes ahalf, same k order as A

  const unsigned short* WT   = (mtx == 0) ? WqT : (mtx == 1) ? WkT : WvT;
  const float*          bias = (mtx == 0) ? bq  : (mtx == 1) ? bk  : bv;

  const unsigned short* Arow   = xb + (size_t)(b * SEQ + Mbase + ln16) * DM;
  const unsigned short* Wgbase = WT + ((size_t)h * DM + Nbase) * DM;

  fill_tile(Wgbase, DM, Bbuf[0], threadIdx.x);

  f32x8 acc[4] = {};
  for (int k = 0; k < DM; k += 32) {
    const int cur = (k >> 5) & 1;
    if (k + 32 < DM) {
      fill_tile(Wgbase + k + 32, DM, Bbuf[cur ^ 1], threadIdx.x);
      wait_async2();
    } else {
      wait_async0();
    }
    __syncthreads();
    __builtin_prefetch(Arow + k + 128, 0, 0);
    // load A + all four B fragments into distinct registers, then issue the
    // wmma chain back-to-back (no per-op dscnt stall)
    bf16x16 a = load2(Arow + k + ahalf, Arow + k + 16 + ahalf);
    bf16x16 bbv[4];
    #pragma unroll
    for (int t = 0; t < 4; t++)
      bbv[t] = load2(&Bbuf[cur][t * 16 + ln16][ahalf],
                     &Bbuf[cur][t * 16 + ln16][16 + ahalf]);
    #pragma unroll
    for (int t = 0; t < 4; t++)
      acc[t] = __builtin_amdgcn_wmma_f32_16x16x32_bf16(
          false, a, false, bbv[t], (short)0, acc[t], false, false);
    __syncthreads();
  }

  const size_t headbase = (size_t)bh * SEQ;
  #pragma unroll
  for (int t = 0; t < 4; t++) {
    const int col = Nbase + t * 16 + ln16;
    const float bv_ = bias[h * DM + col];
    if (mtx == 2) {
      // store transposed: Vt[bh][e=col][s]; lane's 8 rows are contiguous in s
      unsigned int pk[4];
      #pragma unroll
      for (int j = 0; j < 4; j++)
        pk[j] = pack2bf(acc[t][2 * j] + bv_, acc[t][2 * j + 1] + bv_);
      *(uint4*)(Vt + ((size_t)bh * DM + col) * SEQ + Mbase + ahalf) =
          make_uint4(pk[0], pk[1], pk[2], pk[3]);
    } else {
      unsigned short* dst = (mtx == 0) ? Qb : Kb;
      const float sc = (mtx == 0) ? SCALE : 1.0f;
      #pragma unroll
      for (int i = 0; i < 8; i++) {
        const int row = Mbase + i + ahalf;
        dst[(headbase + row) * DM + col] = f2bf((acc[t][i] + bv_) * sc);
      }
    }
  }
}

// ---------------------------------------------------------------------------
// K3: flash attention.  block = 128 thr (4 waves) owns one 16-row q tile of
// one (b,h).  Waves split score depth (128 each) -> LDS reduce -> wave0 online
// softmax -> every wave does P·V on its 128-column slice of Vt.
// Q fragments are register-resident across the whole kv loop.
// grid: (SEQ/16, BATCH*NH)
// ---------------------------------------------------------------------------
__global__ __launch_bounds__(128)
void k_attn(const unsigned short* __restrict__ Qb,
            const unsigned short* __restrict__ Kb,
            const unsigned short* __restrict__ Vt,
            unsigned short* __restrict__ Cc) {
  __shared__ float Spart[4][16][32];
  __shared__ __align__(16) unsigned short Pb[16][32];
  __shared__ float mrow[16], lrow[16], crow[16], linv[16];

  const int lane  = threadIdx.x & 31;
  const int wv    = threadIdx.x >> 5;
  const int ln16  = lane & 15;
  const int hi    = lane >> 4;
  const int ahalf = hi * 8;
  const int bhalf = hi * 16;
  const int qBase = blockIdx.x * 16;
  const int bh    = blockIdx.y;
  const int b     = bh / NH;
  const int h     = bh % NH;
  const size_t hb = (size_t)bh * SEQ;
  const int eBase = wv * 128;

  if (threadIdx.x < 16) { mrow[threadIdx.x] = -3.0e38f; lrow[threadIdx.x] = 0.0f; }

  // hoist this wave's 16x128 Q depth-slice into registers (4 fragments)
  const unsigned short* Arow = Qb + (hb + qBase + ln16) * DM;
  bf16x16 qa[4];
  #pragma unroll
  for (int j = 0; j < 4; j++) {
    const int kd = wv * 128 + j * 32;
    qa[j] = load2(Arow + kd + ahalf, Arow + kd + 16 + ahalf);
  }

  f32x8 o[8] = {};

  for (int kv0 = 0; kv0 < SEQ; kv0 += 32) {
    __syncthreads();   // stats ready / previous iteration's P+corr consumed

    // ---- partial scores over depth [wv*128, wv*128+128) ----
    f32x8 sc[2] = {};
    #pragma unroll
    for (int j = 0; j < 4; j++) {
      const int kd = wv * 128 + j * 32;
      const unsigned short* Br0 = Kb + (hb + kv0 + ln16) * DM + kd + ahalf;
      const unsigned short* Br1 = Br0 + (size_t)16 * DM;
      bf16x16 b0 = load2(Br0, Br0 + 16);
      bf16x16 b1 = load2(Br1, Br1 + 16);
      sc[0] = __builtin_amdgcn_wmma_f32_16x16x32_bf16(
          false, qa[j], false, b0, (short)0, sc[0], false, false);
      sc[1] = __builtin_amdgcn_wmma_f32_16x16x32_bf16(
          false, qa[j], false, b1, (short)0, sc[1], false, false);
    }
    #pragma unroll
    for (int t = 0; t < 2; t++)
      #pragma unroll
      for (int i = 0; i < 8; i++)
        Spart[wv][i + ahalf][t * 16 + ln16] = sc[t][i];
    __syncthreads();

    // ---- wave 0: reduce partials + online softmax, emit bf16 P tile ----
    if (wv == 0) {
      const int r = ln16, ch = hi;
      float ev[16];
      float mx = -3.0e38f;
      #pragma unroll
      for (int j = 0; j < 16; j++) {
        float s = Spart[0][r][ch * 16 + j] + Spart[1][r][ch * 16 + j] +
                  Spart[2][r][ch * 16 + j] + Spart[3][r][ch * 16 + j];
        ev[j] = s;
        mx = fmaxf(mx, s);
      }
      mx = fmaxf(mx, __shfl_xor(mx, 16, 32));
      const float mold = mrow[r];
      const float mnew = fmaxf(mold, mx);
      float ls = 0.0f;
      #pragma unroll
      for (int j = 0; j < 16; j++) { ev[j] = __expf(ev[j] - mnew); ls += ev[j]; }
      ls += __shfl_xor(ls, 16, 32);
      const float cr = __expf(mold - mnew);
      if (ch == 0) { mrow[r] = mnew; lrow[r] = lrow[r] * cr + ls; crow[r] = cr; }
      unsigned int pk[8];
      #pragma unroll
      for (int j = 0; j < 8; j++) pk[j] = pack2bf(ev[2 * j], ev[2 * j + 1]);
      uint4* pdst = (uint4*)&Pb[r][ch * 16];
      pdst[0] = make_uint4(pk[0], pk[1], pk[2], pk[3]);
      pdst[1] = make_uint4(pk[4], pk[5], pk[6], pk[7]);
    }
    __syncthreads();

    // ---- all waves: rescale O, accumulate P·V on 128-col slice of Vt ----
    float cr8[8];
    #pragma unroll
    for (int i = 0; i < 8; i++) cr8[i] = crow[i + ahalf];
    #pragma unroll
    for (int t = 0; t < 8; t++)
      #pragma unroll
      for (int i = 0; i < 8; i++) o[t][i] *= cr8[i];

    bf16x16 pa = load2(&Pb[ln16][ahalf], &Pb[ln16][16 + ahalf]);
    const unsigned short* Vbase =
        Vt + ((size_t)bh * DM + eBase + ln16) * SEQ + kv0 + ahalf;
    #pragma unroll
    for (int t = 0; t < 8; t += 2) {
      const unsigned short* Vr0 = Vbase + (size_t)(t)     * 16 * SEQ;
      const unsigned short* Vr1 = Vbase + (size_t)(t + 1) * 16 * SEQ;
      bf16x16 b0 = load2(Vr0, Vr0 + 16);
      bf16x16 b1 = load2(Vr1, Vr1 + 16);
      o[t] = __builtin_amdgcn_wmma_f32_16x16x32_bf16(
          false, pa, false, b0, (short)0, o[t], false, false);
      o[t + 1] = __builtin_amdgcn_wmma_f32_16x16x32_bf16(
          false, pa, false, b1, (short)0, o[t + 1], false, false);
    }
  }

  __syncthreads();
  if (threadIdx.x < 16) linv[threadIdx.x] = 1.0f / lrow[threadIdx.x];
  __syncthreads();

  float li8[8];
  #pragma unroll
  for (int i = 0; i < 8; i++) li8[i] = linv[i + ahalf];
  #pragma unroll
  for (int t = 0; t < 8; t++) {
    const int col = h * DM + eBase + t * 16 + ln16;
    #pragma unroll
    for (int i = 0; i < 8; i++) {
      const int row = qBase + i + ahalf;
      Cc[((size_t)b * SEQ + row) * HD + col] = f2bf(o[t][i] * li8[i]);
    }
  }
}

// ---------------------------------------------------------------------------
// K4: output projection + bias + residual.  Same async-staged LDS weight tile
// as K2.  wave = 16x64 strip, K over 4096.
// grid: (BATCH*SEQ/64, DM/64)
// ---------------------------------------------------------------------------
__global__ __launch_bounds__(128)
void k_outproj(const unsigned short* __restrict__ Cc,
               const unsigned short* __restrict__ WoT,
               const float* __restrict__ bo,
               const float* __restrict__ x,
               float* __restrict__ out) {
  __shared__ __align__(16) unsigned short Bbuf[2][64][32];

  const int lane  = threadIdx.x & 31;
  const int wv    = threadIdx.x >> 5;
  const int ln16  = lane & 15;
  const int hi    = lane >> 4;
  const int ahalf = hi * 8;
  const int bhalf = hi * 16;
  const int Mbase = blockIdx.x * 64 + wv * 16;   // row into B*S
  const int Nbase = blockIdx.y * 64;

  const unsigned short* Arow   = Cc + (size_t)(Mbase + ln16) * HD;
  const unsigned short* Wgbase = WoT + (size_t)Nbase * HD;

  fill_tile(Wgbase, HD, Bbuf[0], threadIdx.x);

  f32x8 acc[4] = {};
  for (int k = 0; k < HD; k += 32) {
    const int cur = (k >> 5) & 1;
    if (k + 32 < HD) {
      fill_tile(Wgbase + k + 32, HD, Bbuf[cur ^ 1], threadIdx.x);
      wait_async2();
    } else {
      wait_async0();
    }
    __syncthreads();
    __builtin_prefetch(Arow + k + 128, 0, 0);
    bf16x16 a = load2(Arow + k + ahalf, Arow + k + 16 + ahalf);
    bf16x16 bbv[4];
    #pragma unroll
    for (int t = 0; t < 4; t++)
      bbv[t] = load2(&Bbuf[cur][t * 16 + ln16][ahalf],
                     &Bbuf[cur][t * 16 + ln16][16 + ahalf]);
    #pragma unroll
    for (int t = 0; t < 4; t++)
      acc[t] = __builtin_amdgcn_wmma_f32_16x16x32_bf16(
          false, a, false, bbv[t], (short)0, acc[t], false, false);
    __syncthreads();
  }

  #pragma unroll
  for (int t = 0; t < 4; t++) {
    const int col = Nbase + t * 16 + ln16;
    const float bb = bo[col];
    #pragma unroll
    for (int i = 0; i < 8; i++) {
      const size_t row = (size_t)Mbase + i + ahalf;
      out[row * DM + col] = acc[t][i] + bb + x[row * DM + col];
    }
  }
}

// ---------------------------------------------------------------------------
// Launch
// ---------------------------------------------------------------------------
extern "C" void kernel_launch(void* const* d_in, const int* in_sizes, int n_in,
                              void* d_out, int out_size, void* d_ws, size_t ws_size,
                              hipStream_t stream) {
  const float* x  = (const float*)d_in[0];
  const float* Wq = (const float*)d_in[1];
  const float* Wk = (const float*)d_in[2];
  const float* Wv = (const float*)d_in[3];
  const float* bq = (const float*)d_in[4];
  const float* bk = (const float*)d_in[5];
  const float* bv = (const float*)d_in[6];
  const float* Wo = (const float*)d_in[7];
  const float* bo = (const float*)d_in[8];
  float* out = (float*)d_out;

  // workspace layout (bf16 / ushort elements)
  unsigned short* p   = (unsigned short*)d_ws;
  unsigned short* xb  = p;  p += (size_t)BATCH * SEQ * DM;   //  4 MB
  unsigned short* WqT = p;  p += (size_t)NH * DM * DM;       //  4 MB
  unsigned short* WkT = p;  p += (size_t)NH * DM * DM;       //  4 MB
  unsigned short* WvT = p;  p += (size_t)NH * DM * DM;       //  4 MB
  unsigned short* WoT = p;  p += (size_t)DM * HD;            //  4 MB
  unsigned short* Qb  = p;  p += (size_t)BATCH * NH * SEQ * DM;  // 32 MB
  unsigned short* Kb  = p;  p += (size_t)BATCH * NH * SEQ * DM;  // 32 MB
  unsigned short* Vt  = p;  p += (size_t)BATCH * NH * DM * SEQ;  // 32 MB
  unsigned short* Cc  = p;  p += (size_t)BATCH * SEQ * HD;       // 32 MB
  size_t need = (size_t)((char*)p - (char*)d_ws);
  if (ws_size < need) return;   // scratch too small: bail deterministically

  const int nx = BATCH * SEQ * DM;
  k_cvt_x<<<(nx + 255) / 256, 256, 0, stream>>>(x, xb, nx);

  dim3 tb(32, 8);
  k_transpose_bf16<<<dim3(16, 16, NH), tb, 0, stream>>>(Wq, WqT, DM, DM);
  k_transpose_bf16<<<dim3(16, 16, NH), tb, 0, stream>>>(Wk, WkT, DM, DM);
  k_transpose_bf16<<<dim3(16, 16, NH), tb, 0, stream>>>(Wv, WvT, DM, DM);
  k_transpose_bf16<<<dim3(16, 128, 1), tb, 0, stream>>>(Wo, WoT, HD, DM);

  k_qkv_gemm<<<dim3(SEQ / 64, DM / 64, 3 * BATCH * NH), 128, 0, stream>>>(
      xb, WqT, WkT, WvT, bq, bk, bv, Qb, Kb, Vt);

  k_attn<<<dim3(SEQ / 16, BATCH * NH), 128, 0, stream>>>(Qb, Kb, Vt, Cc);

  k_outproj<<<dim3(BATCH * SEQ / 64, DM / 64), 128, 0, stream>>>(Cc, WoT, bo, x, out);
}
